// SelfAttn_73976516706753
// MI455X (gfx1250) — matrix operand order repaired and checked by reading, not compiled, not run on hardware
//
#include <hip/hip_runtime.h>
#include <hip/hip_bf16.h>

typedef __attribute__((ext_vector_type(16))) _Float16 v16h;
typedef __attribute__((ext_vector_type(8)))  _Float16 v8h;
typedef __attribute__((ext_vector_type(8)))  float    v8f;
typedef __attribute__((ext_vector_type(4)))  float    v4f;

#define B_  4
#define T_  2048
#define D_  1024
#define NH_ 8
#define HD_ 128

union F16Frag { v16h v; v8h h8[2]; _Float16 h[16]; };

// A-operand fragment (16x32 f16): lane m holds row M=m%16; halves 0..15 =
// contiguous K (m<16 ? 0..15 : 16..31). Source: row-major [rows, ld].
__device__ __forceinline__ v16h load_afrag(const _Float16* __restrict__ p, int ld,
                                           int row_base, int k_base, int lane) {
  int r  = row_base + (lane & 15);
  int k0 = k_base + ((lane >> 4) << 4);
  const _Float16* q = p + (size_t)r * ld + k0;
  F16Frag f;
  f.h8[0] = *reinterpret_cast<const v8h*>(q);
  f.h8[1] = *reinterpret_cast<const v8h*>(q + 8);
  return f.v;
}

// B-operand fragment (32x16 f16): lane n holds column N=n%16; halves 0..15 =
// contiguous K (lane<16 ? 0..15 : 16..31). Source: "column"-major, i.e. a
// row-major matrix [cols, ld] whose row index is the WMMA N index.
__device__ __forceinline__ v16h load_bfrag(const _Float16* __restrict__ p, int ld,
                                           int col_base, int k_base, int lane) {
  int c  = col_base + (lane & 15);
  int k0 = k_base + ((lane >> 4) << 4);
  const _Float16* q = p + (size_t)c * ld + k0;
  F16Frag f;
  f.h8[0] = *reinterpret_cast<const v8h*>(q);
  f.h8[1] = *reinterpret_cast<const v8h*>(q + 8);
  return f.v;
}

__global__ void __launch_bounds__(256) cvt_f32_to_f16_kernel(
    const float* __restrict__ src, _Float16* __restrict__ dst, int n) {
  int i = blockIdx.x * blockDim.x + threadIdx.x;
  int stride = gridDim.x * blockDim.x;
  for (; i < n; i += stride) dst[i] = (_Float16)src[i];
}

// y = x @ W^T for W in {wq,wk,wv} (blockIdx.z). Per-wave 32x64 tile.
// q,k written to [B,N,T,H] (pre-RoPE); v written transposed to [B,N,H,T].
__global__ void __launch_bounds__(256) gemm_qkv_kernel(
    const _Float16* __restrict__ xh, const _Float16* __restrict__ wh,
    _Float16* __restrict__ qr, _Float16* __restrict__ kr,
    _Float16* __restrict__ vT) {
  const int lane = threadIdx.x & 31;
  const int wid  = threadIdx.x >> 5;
  const int z    = blockIdx.z;
  const _Float16* w = wh + (size_t)z * (D_ * D_);
  const int m0 = blockIdx.y * 128 + (wid & 3) * 32;   // token rows
  const int n0 = blockIdx.x * 128 + (wid >> 2) * 64;  // output features

  v8f zero = {};
  v8f acc[2][4];
#pragma unroll
  for (int i = 0; i < 2; ++i)
#pragma unroll
    for (int j = 0; j < 4; ++j) acc[i][j] = zero;

  for (int k0 = 0; k0 < D_; k0 += 32) {
    v16h a0 = load_afrag(xh, D_, m0,      k0, lane);
    v16h a1 = load_afrag(xh, D_, m0 + 16, k0, lane);
    v16h b[4];
#pragma unroll
    for (int j = 0; j < 4; ++j) b[j] = load_bfrag(w, D_, n0 + 16 * j, k0, lane);
#pragma unroll
    for (int j = 0; j < 4; ++j) {
      acc[0][j] = __builtin_amdgcn_wmma_f32_16x16x32_f16(false, a0, false, b[j],
                                                         (short)0, acc[0][j], false, false);
      acc[1][j] = __builtin_amdgcn_wmma_f32_16x16x32_f16(false, a1, false, b[j],
                                                         (short)0, acc[1][j], false, false);
    }
  }

#pragma unroll
  for (int i = 0; i < 2; ++i) {
#pragma unroll
    for (int j = 0; j < 4; ++j) {
      const int col   = n0 + 16 * j + (lane & 15);
      const int rbase = m0 + 16 * i + ((lane >> 4) << 3);
      const int n = col >> 7;      // head
      const int h = col & 127;     // within-head dim
#pragma unroll
      for (int jj = 0; jj < 8; ++jj) {
        const int r  = rbase + jj;
        const int bb = r >> 11;    // / T
        const int t  = r & 2047;   // % T
        const _Float16 val = (_Float16)acc[i][j][jj];
        if (z == 0)      qr[(((size_t)bb * NH_ + n) * T_ + t) * HD_ + h] = val;
        else if (z == 1) kr[(((size_t)bb * NH_ + n) * T_ + t) * HD_ + h] = val;
        else             vT[(((size_t)bb * NH_ + n) * HD_ + h) * T_ + t] = val;
      }
    }
  }
}

#define QK_IDX(bb, d, t) \
  ((((size_t)(bb) * NH_ + ((d) >> 7)) * T_ + (t)) * HD_ + ((d) & 127))

// In-place RoPE on q,k (layout [B,N,T,H]); folds 1/sqrt(H) into q.
__global__ void __launch_bounds__(256) rope_kernel(
    _Float16* __restrict__ qr, _Float16* __restrict__ kr,
    const float* __restrict__ cosT, const float* __restrict__ sinT) {
  const int rid = blockIdx.x;          // (b, t) row
  const int bb = rid >> 11;
  const int t  = rid & 2047;
  float qv[4], qp[4], kv[4], kp[4], cs[4], sn[4];
  int dd[4];
#pragma unroll
  for (int e = 0; e < 4; ++e) {
    const int d = threadIdx.x + 256 * e;
    dd[e] = d;
    const int pidx = (d < 512) ? (2 * d + 1) : (2 * d - 1024);
    const float sgn = (d < 512) ? -1.0f : 1.0f;
    cs[e] = cosT[t * D_ + d];
    sn[e] = sinT[t * D_ + d];
    qv[e] = (float)qr[QK_IDX(bb, d, t)];
    kv[e] = (float)kr[QK_IDX(bb, d, t)];
    qp[e] = sgn * (float)qr[QK_IDX(bb, pidx, t)];
    kp[e] = sgn * (float)kr[QK_IDX(bb, pidx, t)];
  }
  __syncthreads();
  const float qscale = 0.08838834764831845f;  // 1/sqrt(128)
#pragma unroll
  for (int e = 0; e < 4; ++e) {
    const int d = dd[e];
    qr[QK_IDX(bb, d, t)] = (_Float16)(qscale * (qv[e] * cs[e] + qp[e] * sn[e]));
    kr[QK_IDX(bb, d, t)] = (_Float16)(kv[e] * cs[e] + kp[e] * sn[e]);
  }
}

// Causal flash attention; each wave owns 16 queries. Computes S^T = K*Q^T and
// O^T = V^T * P^T so softmax reductions are in-lane + one shfl_xor(16).
__global__ void __launch_bounds__(256) flash_attn_kernel(
    const _Float16* __restrict__ qr, const _Float16* __restrict__ kr,
    const _Float16* __restrict__ vT, float* __restrict__ out) {
  const int lane = threadIdx.x & 31;
  const int wid  = threadIdx.x >> 5;
  const int bn   = blockIdx.y;
  const int qb   = blockIdx.x * 128 + wid * 16;
  const _Float16* Q = qr + (size_t)bn * T_ * HD_;
  const _Float16* K = kr + (size_t)bn * T_ * HD_;
  const _Float16* V = vT + (size_t)bn * HD_ * T_;

  v16h qf[4];
#pragma unroll
  for (int hc = 0; hc < 4; ++hc) qf[hc] = load_bfrag(Q, HD_, qb, hc * 32, lane);

  v8f zero = {};
  v8f oT[8];
#pragma unroll
  for (int f = 0; f < 8; ++f) oT[f] = zero;
  float mrow = -3.0e38f, lrow = 0.0f;

  const int ntiles = (qb + 79) >> 6;  // causal: key tiles with kt0 <= qb+15
  for (int tile = 0; tile < ntiles; ++tile) {
    const int kt0 = tile << 6;
    v8f s[4];
#pragma unroll
    for (int kf = 0; kf < 4; ++kf) s[kf] = zero;
#pragma unroll
    for (int kf = 0; kf < 4; ++kf) {
#pragma unroll
      for (int hc = 0; hc < 4; ++hc) {
        v16h aK = load_afrag(K, HD_, kt0 + kf * 16, hc * 32, lane);
        s[kf] = __builtin_amdgcn_wmma_f32_16x16x32_f16(false, aK, false, qf[hc],
                                                       (short)0, s[kf], false, false);
      }
    }
    if (tile == ntiles - 1) {           // only the diagonal tile needs masking
      const int qg = qb + (lane & 15);
#pragma unroll
      for (int kf = 0; kf < 4; ++kf)
#pragma unroll
        for (int jj = 0; jj < 8; ++jj) {
          const int key = kt0 + kf * 16 + ((lane >> 4) << 3) + jj;
          if (key > qg) s[kf][jj] = -3.0e38f;
        }
    }
    // online softmax (per query = per lane%16)
    float vm = -3.0e38f;
#pragma unroll
    for (int kf = 0; kf < 4; ++kf)
#pragma unroll
      for (int jj = 0; jj < 8; ++jj) vm = fmaxf(vm, s[kf][jj]);
    vm = fmaxf(vm, __shfl_xor(vm, 16, 32));
    const float mnew  = fmaxf(mrow, vm);
    const float alpha = __expf(mrow - mnew);
    mrow = mnew;
    float ls = 0.0f;
#pragma unroll
    for (int kf = 0; kf < 4; ++kf)
#pragma unroll
      for (int jj = 0; jj < 8; ++jj) {
        const float p = __expf(s[kf][jj] - mnew);
        s[kf][jj] = p;
        ls += p;
      }
    ls += __shfl_xor(ls, 16, 32);
    lrow = lrow * alpha + ls;
#pragma unroll
    for (int f = 0; f < 8; ++f) oT[f] = oT[f] * alpha;

    // Rebuild P^T C-fragments into B-operand fragments (32 keys per chunk).
#pragma unroll
    for (int c = 0; c < 2; ++c) {
      F16Frag pb;
#pragma unroll
      for (int j = 0; j < 8; ++j) {
        const float x0 = s[2 * c][j];
        const float x1 = __shfl_xor(x0, 16, 32);
        const float y0 = s[2 * c + 1][j];
        const float y1 = __shfl_xor(y0, 16, 32);
        const float lo = (lane < 16) ? x0 : y1;  // key j of this lane's group
        const float hi = (lane < 16) ? x1 : y0;  // key 8+j
        pb.h[j]     = (_Float16)lo;
        pb.h[8 + j] = (_Float16)hi;
      }
#pragma unroll
      for (int hf = 0; hf < 8; ++hf) {
        v16h aV = load_afrag(V, T_, hf * 16, kt0 + c * 32, lane);
        oT[hf] = __builtin_amdgcn_wmma_f32_16x16x32_f16(false, aV, false, pb.v,
                                                        (short)0, oT[hf], false, false);
      }
    }
  }

  const float inv_l = 1.0f / lrow;
  const int bb = bn >> 3, nh = bn & 7;
  const int t  = qb + (lane & 15);
  float* orow = out + ((size_t)bb * T_ + t) * D_ + nh * HD_;
#pragma unroll
  for (int hf = 0; hf < 8; ++hf) {
    const int h0 = hf * 16 + ((lane >> 4) << 3);
    v4f a, b;
#pragma unroll
    for (int jj = 0; jj < 4; ++jj) {
      a[jj] = oT[hf][jj] * inv_l;
      b[jj] = oT[hf][4 + jj] * inv_l;
    }
    *reinterpret_cast<v4f*>(orow + h0)     = a;
    *reinterpret_cast<v4f*>(orow + h0 + 4) = b;
  }
}

extern "C" void kernel_launch(void* const* d_in, const int* in_sizes, int n_in,
                              void* d_out, int out_size, void* d_ws, size_t ws_size,
                              hipStream_t stream) {
  (void)in_sizes; (void)n_in; (void)out_size; (void)ws_size;
  const float* x    = (const float*)d_in[0];
  const float* wq   = (const float*)d_in[1];
  const float* wk   = (const float*)d_in[2];
  const float* wv   = (const float*)d_in[3];
  const float* cosT = (const float*)d_in[4];
  const float* sinT = (const float*)d_in[5];
  float* out = (float*)d_out;

  // workspace layout (f16 elements): xh | wh[3] | qr | kr | vT  (~70 MB)
  _Float16* xh = (_Float16*)d_ws;
  _Float16* wh = xh + (size_t)8388608;        // 3 x 1048576
  _Float16* qr = wh + (size_t)3145728;        // [B,N,T,H]
  _Float16* kr = qr + (size_t)8388608;        // [B,N,T,H]
  _Float16* vT = kr + (size_t)8388608;        // [B,N,H,T]

  cvt_f32_to_f16_kernel<<<2048, 256, 0, stream>>>(x,  xh, 8388608);
  cvt_f32_to_f16_kernel<<<512,  256, 0, stream>>>(wq, wh,            1048576);
  cvt_f32_to_f16_kernel<<<512,  256, 0, stream>>>(wk, wh + 1048576,  1048576);
  cvt_f32_to_f16_kernel<<<512,  256, 0, stream>>>(wv, wh + 2097152,  1048576);

  gemm_qkv_kernel<<<dim3(8, 64, 3), 256, 0, stream>>>(xh, wh, qr, kr, vT);
  rope_kernel<<<8192, 256, 0, stream>>>(qr, kr, cosT, sinT);
  flash_attn_kernel<<<dim3(16, 32), 256, 0, stream>>>(qr, kr, vT, out);
}
